// WaveDensityBlock_17051020165575
// MI455X (gfx1250) — matrix operand order repaired and checked by reading, not compiled, not run on hardware
//
#include <hip/hip_runtime.h>
#include <hip/hip_bf16.h>

typedef __attribute__((ext_vector_type(16))) _Float16 v16h;
typedef __attribute__((ext_vector_type(8)))  float    v8f;
typedef __attribute__((ext_vector_type(4)))  unsigned int v4u;
typedef __attribute__((ext_vector_type(8)))  int      v8i;
typedef __attribute__((ext_vector_type(4)))  int      v4i;

#define B_    4
#define S_    4096
#define D_    1024
#define H_    8
#define M_    32
#define W_    16
#define NS_   128
#define GH_   128
#define HD_   128
#define TEMP_ 0.6f
#define SIGMA_ 0.12f
#define ALPHA_ 6.0f
#define PI_   3.14159265358979323846f

#if __has_builtin(__builtin_amdgcn_tensor_load_to_lds)
#define HAVE_TDM 1
#else
#define HAVE_TDM 0
#endif

// LDS row stride for 64-half tile rows with TDM pad (32+2 DWORDs)
#define LSTRIDE 68

// ---------------------------------------------------------------- zero init
__global__ void k_zero(float* __restrict__ pooled, float* __restrict__ dsum) {
    int i = blockIdx.x * 256 + threadIdx.x;
    if (i < B_ * D_) pooled[i] = 0.0f;
    if (i < H_ * B_) dsum[i] = 0.0f;
}

// ------------------------------------------------- pooled = mean over S of x
__global__ void __launch_bounds__(256) k_pool(const float* __restrict__ x,
                                              float* __restrict__ pooled) {
    int bd = blockIdx.x * 256 + threadIdx.x;      // 0..B*D-1
    int b = bd >> 10, d = bd & (D_ - 1);
    int s0 = blockIdx.y * 256;
    const float* px = x + ((size_t)b * S_ + s0) * D_ + d;
    float acc = 0.0f;
    for (int s = 0; s < 256; ++s) acc += px[(size_t)s * D_];
    atomicAdd(&pooled[bd], acc * (1.0f / S_));
}

// ---------------------------------- gating MLP + softmax -> weights[H,B,M]
// forward value of straight-through topk == soft (hard cancels), so softmax only.
__global__ void __launch_bounds__(128) k_gate(const float* __restrict__ pooled,
                                              const float* __restrict__ W1,
                                              const float* __restrict__ b1,
                                              const float* __restrict__ W2,
                                              const float* __restrict__ b2,
                                              float* __restrict__ weights) {
    int h = blockIdx.x >> 2;
    int b = blockIdx.x & 3;
    __shared__ float hpre[GH_];
    __shared__ float lg[M_];
    __shared__ float invsum;
    int g = threadIdx.x;
    const float* p  = pooled + b * D_;
    const float* w1 = W1 + (size_t)h * D_ * GH_ + g;
    float acc = b1[h * GH_ + g];
    for (int d = 0; d < D_; ++d) acc += p[d] * w1[(size_t)d * GH_];
    hpre[g] = fmaxf(acc, 0.0f);
    __syncthreads();
    if (threadIdx.x < M_) {
        int m = threadIdx.x;
        const float* w2 = W2 + (size_t)h * GH_ * M_ + m;
        float l = b2[h * M_ + m];
        for (int gg = 0; gg < GH_; ++gg) l += hpre[gg] * w2[gg * M_];
        lg[m] = l * (1.0f / TEMP_);
    }
    __syncthreads();
    if (threadIdx.x == 0) {
        float mx = lg[0];
        for (int m = 1; m < M_; ++m) mx = fmaxf(mx, lg[m]);
        float sum = 0.0f;
        for (int m = 0; m < M_; ++m) { float e = __expf(lg[m] - mx); lg[m] = e; sum += e; }
        invsum = 1.0f / sum;
    }
    __syncthreads();
    if (threadIdx.x < M_)
        weights[(h * B_ + b) * M_ + threadIdx.x] = lg[threadIdx.x] * invsum;
}

// ------------------------------------------- mask_grids[H,M,S] from waves
__global__ void __launch_bounds__(256) k_mask(const float* __restrict__ freqs,
                                              const float* __restrict__ amps,
                                              const float* __restrict__ phases,
                                              float* __restrict__ mask) {
    int idx = blockIdx.x * 256 + threadIdx.x;   // H*M*S
    int s  = idx & (S_ - 1);
    int hm = idx >> 12;
    int sx = s & 63, sy = s >> 6;               // meshgrid indexing='xy'
    float px = -1.0f + (2.0f / 63.0f) * sx;
    float py = -1.0f + (2.0f / 63.0f) * sy;
    const float* f  = freqs  + (size_t)hm * W_ * 2;
    const float* a  = amps   + (size_t)hm * W_;
    const float* ph = phases + (size_t)hm * W_;
    float acc = 0.0f;
    #pragma unroll
    for (int w = 0; w < W_; ++w) {
        float dot = px * f[2 * w] + py * f[2 * w + 1];
        float arg = 2.0f * PI_ * dot + ph[w];
        float t = (arg - 0.5f * PI_) * (1.0f / (2.0f * PI_));
        float frac = t - floorf(t);
        float tri01 = 1.0f - 2.0f * fabsf(frac - 0.5f);
        acc += (2.0f * tri01 - 1.0f) * a[w];
    }
    mask[idx] = acc;
}

// ------------- density[H,B,S] (unnormalized) + per-(h,b) sums via atomics.
// forward value of straight-through hits == sigmoid(ALPHA*samples).
__global__ void __launch_bounds__(256) k_density(const float* __restrict__ mask,
                                                 const float* __restrict__ weights,
                                                 const float* __restrict__ noise,
                                                 float* __restrict__ draw,
                                                 float* __restrict__ dsum) {
    int h = blockIdx.z, b = blockIdx.y;
    int s = blockIdx.x * 256 + threadIdx.x;
    __shared__ float wsh[M_];
    __shared__ float red[256];
    if (threadIdx.x < M_) wsh[threadIdx.x] = weights[(h * B_ + b) * M_ + threadIdx.x];
    __syncthreads();
    float fg = 0.0f;
    const float* mg = mask + (size_t)h * M_ * S_ + s;
    #pragma unroll 8
    for (int m = 0; m < M_; ++m) fg += wsh[m] * mg[(size_t)m * S_];
    const float* nz = noise + ((size_t)(h * B_ + b) * NS_) * S_ + s;
    float dens = 0.0f;
    for (int n = 0; n < NS_; ++n) {
        float v = fg + SIGMA_ * nz[(size_t)n * S_];
        dens += 1.0f / (1.0f + __expf(-ALPHA_ * v));
    }
    dens *= (1.0f / NS_);
    draw[(size_t)(h * B_ + b) * S_ + s] = dens;
    red[threadIdx.x] = dens;
    __syncthreads();
    for (int off = 128; off > 0; off >>= 1) {
        if (threadIdx.x < off) red[threadIdx.x] += red[threadIdx.x + off];
        __syncthreads();
    }
    if (threadIdx.x == 0) atomicAdd(&dsum[h * B_ + b], red[0]);
}

// ------------- mh_f16[b*S+s][d] = normalized_density * x  (f16 for WMMA A)
__global__ void __launch_bounds__(256) k_modulate(const float* __restrict__ x,
                                                  const float* __restrict__ draw,
                                                  const float* __restrict__ dsum,
                                                  _Float16* __restrict__ mh) {
    size_t idx = (size_t)blockIdx.x * 256 + threadIdx.x;  // B*S*D
    int d = (int)(idx & (D_ - 1));
    size_t row = idx >> 10;                 // b*S + s
    int b = (int)(row >> 12);
    int s = (int)(row & (S_ - 1));
    int h = d >> 7;                         // HD = 128
    int hb = h * B_ + b;
    float dn = draw[(size_t)hb * S_ + s] / (dsum[hb] + 1e-8f);
    mh[idx] = (_Float16)(dn * x[idx]);
}

// --------------------- Wo f32 -> f16, transposed: wofT[n][k] = Wo[k][n]
// K-contiguous B-operand storage so LDS fragment gathers are contiguous.
__global__ void __launch_bounds__(256) k_wo16t(const float* __restrict__ Wo,
                                               _Float16* __restrict__ wofT) {
    __shared__ float tile[32][33];
    int k0 = blockIdx.x * 32, n0 = blockIdx.y * 32;
    int tx = threadIdx.x & 31, ty = threadIdx.x >> 5;   // 32 x 8
    #pragma unroll
    for (int i = ty; i < 32; i += 8)
        tile[i][tx] = Wo[(size_t)(k0 + i) * D_ + n0 + tx];
    __syncthreads();
    #pragma unroll
    for (int i = ty; i < 32; i += 8)
        wofT[(size_t)(n0 + i) * D_ + k0 + tx] = (_Float16)tile[tx][i];
}

// --------------------------------------------------------------- TDM helper
__device__ __forceinline__ unsigned lds_addr_of(const void* p) {
    // low 32 bits of a generic pointer to LDS == byte offset in LDS aperture
    return (unsigned)(unsigned long long)p;
}

#if HAVE_TDM
// D# (ISA ch.8) for a 2D tile load: data_size=2B, pad_enable: +2 DWORDs of
// LDS padding per pad-interval so tile rows land at stride LSTRIDE halfs.
__device__ __forceinline__ void tdm_load_2d(unsigned lds, const void* gptr,
                                            unsigned dim0, unsigned dim1,
                                            unsigned stride0,
                                            unsigned tile0, unsigned tile1,
                                            unsigned pint, unsigned pamt) {
    unsigned long long ga = (unsigned long long)gptr;
    v4u g0;
    g0[0] = 1u;                                    // count=1 (valid user D#)
    g0[1] = lds;                                   // lds_addr
    g0[2] = (unsigned)(ga & 0xFFFFFFFFu);          // global_addr[31:0]
    g0[3] = (unsigned)((ga >> 32) & 0x01FFFFFFu)   // global_addr[56:32]
          | (2u << 30);                            // type = 2 ("image")
    v8i g1;
    g1[0] = (int)((1u << 16)                       // data_size = 2 bytes
                | (1u << 20)                       // pad_enable
                | (pint << 22)                     // pad_interval (2<<p DWORDs)
                | (pamt << 25));                   // pad_amount ((p+1) DWORDs)
    g1[1] = (int)(dim0 << 16);                     // tensor_dim0[15:0]
    g1[2] = (int)((dim0 >> 16) | (dim1 << 16));    // dim0[31:16] | dim1[15:0]
    g1[3] = (int)((dim1 >> 16) | (tile0 << 16));   // dim1[31:16] | tile_dim0
    g1[4] = (int)(tile1 & 0xFFFFu);                // tile_dim1 (tile_dim2 = 0)
    g1[5] = (int)stride0;                          // tensor_dim0_stride[31:0]
    g1[6] = 0;                                     // stride0[47:32]|dim1_stride
    g1[7] = 0;
    v4i z4 = {};
#if __clang_major__ >= 23
    v8i z8 = {};
    __builtin_amdgcn_tensor_load_to_lds(g0, g1, z4, z4, z8, 0);
#else
    __builtin_amdgcn_tensor_load_to_lds(g0, g1, z4, z4, 0);
#endif
}
#endif

// ---------------- out = mh @ Wo + bo + x  via v_wmma_f32_16x16x32_f16
// Block: 16 rows x 128 cols, 8 waves, one 16x16 C tile each. K step 64 per
// double-buffered TDM load (2 chained WMMAs per barrier pair).
// A tile in LDS: [16 rows][LSTRIDE], row-major K.   (from mh, K-contiguous)
// B tile in LDS: [128 cols][LSTRIDE], K-contiguous. (from wofT = Wo^T)
// Fragment layouts per CDNA5 ISA 7.12.2 (wave32):
//   A (16x32 f16): lane L = row M=L%16; halfs [2v,2v+1]: v<4 -> K=2v+8*hi,
//                  v>=4 -> K=16+2(v-4)+8*hi  (hi = L>=16)
//   B (32x16 f16): lane L = col N=L%16; half e -> K = 8*hi + e (+8 for e>=8)
//   C/D (16x16 f32): VGPR v, lane L -> row M = v + 8*hi, col N = L%16
__global__ void __launch_bounds__(256) k_gemm(const _Float16* __restrict__ mh,
                                              const _Float16* __restrict__ wofT,
                                              const float* __restrict__ x,
                                              const float* __restrict__ bo,
                                              float* __restrict__ out) {
    __shared__ _Float16 As[2][16 * LSTRIDE];
    __shared__ _Float16 Bs[2][128 * LSTRIDE];
    int tid  = threadIdx.x;
    int lane = tid & 31, wave = tid >> 5;
    int r = lane & 15, hig = lane >> 4;
    int row0 = blockIdx.y * 16;
    int col0 = blockIdx.x * 128;
    int n = wave * 16 + r;                 // this lane's output column in tile
    v8f c = {};
#if HAVE_TDM
    if (wave == 0) {   // prologue: tiles for it=0 into buffer 0
        tdm_load_2d(lds_addr_of(&As[0][0]), mh + (size_t)row0 * D_,
                    D_, B_ * S_, D_, 64, 16, 4, 1);
        tdm_load_2d(lds_addr_of(&Bs[0][0]), wofT + (size_t)col0 * D_,
                    D_, D_, D_, 64, 128, 4, 1);
    }
    for (int it = 0; it < D_ / 64; ++it) {
        int cur = it & 1;
        if (wave == 0) {
            if (it + 1 < D_ / 64) {   // prefetch next tiles, then wait for cur
                int k1 = (it + 1) * 64;
                tdm_load_2d(lds_addr_of(&As[cur ^ 1][0]),
                            mh + (size_t)row0 * D_ + k1,
                            D_, B_ * S_, D_, 64, 16, 4, 1);
                tdm_load_2d(lds_addr_of(&Bs[cur ^ 1][0]),
                            wofT + (size_t)col0 * D_ + k1,
                            D_, D_, D_, 64, 128, 4, 1);
                __builtin_amdgcn_s_wait_tensorcnt(2);  // cur tiles complete
            } else {
                __builtin_amdgcn_s_wait_tensorcnt(0);
            }
        }
        __syncthreads();
        const _Float16* Ab = As[cur];
        const _Float16* Bb = Bs[cur];
        #pragma unroll
        for (int j = 0; j < 2; ++j) {          // two K=32 sub-steps
            v16h a, bf;
            #pragma unroll
            for (int v = 0; v < 8; ++v) {
                int kb = (v < 4) ? (2 * v + 8 * hig)
                                 : (16 + 2 * (v - 4) + 8 * hig);
                a[2 * v]     = Ab[r * LSTRIDE + j * 32 + kb];
                a[2 * v + 1] = Ab[r * LSTRIDE + j * 32 + kb + 1];
            }
            #pragma unroll
            for (int e = 0; e < 16; ++e)       // table order: K = j*32+8hi+e (+8 for e>=8)
                bf[e] = Bb[n * LSTRIDE + j * 32 + 8 * hig + e + (e >= 8 ? 8 : 0)];
            c = __builtin_amdgcn_wmma_f32_16x16x32_f16(false, a, false, bf,
                                                       (short)0, c,
                                                       false, false);
        }
        __syncthreads();
    }
#else
    // Fallback: manual global->LDS copies (single buffer), same LDS layout.
    for (int k0 = 0; k0 < D_; k0 += 64) {
        {   // A tile 16x64: 1024 halfs, 4 per thread
            int lin = tid * 4;
            int rr = lin >> 6, kk = lin & 63;
            const _Float16* src = mh + (size_t)(row0 + rr) * D_ + k0 + kk;
            #pragma unroll
            for (int j = 0; j < 4; ++j) As[0][rr * LSTRIDE + kk + j] = src[j];
        }
        {   // B tile 128x64 from wofT: 8192 halfs, 32 per thread
            int rr = tid >> 1, off = (tid & 1) * 32;
            const _Float16* src = wofT + (size_t)(col0 + rr) * D_ + k0 + off;
            #pragma unroll
            for (int j = 0; j < 32; ++j) Bs[0][rr * LSTRIDE + off + j] = src[j];
        }
        __syncthreads();
        #pragma unroll
        for (int j = 0; j < 2; ++j) {
            v16h a, bf;
            #pragma unroll
            for (int v = 0; v < 8; ++v) {
                int kb = (v < 4) ? (2 * v + 8 * hig)
                                 : (16 + 2 * (v - 4) + 8 * hig);
                a[2 * v]     = As[0][r * LSTRIDE + j * 32 + kb];
                a[2 * v + 1] = As[0][r * LSTRIDE + j * 32 + kb + 1];
            }
            #pragma unroll
            for (int e = 0; e < 16; ++e)
                bf[e] = Bs[0][n * LSTRIDE + j * 32 + 8 * hig + e + (e >= 8 ? 8 : 0)];
            c = __builtin_amdgcn_wmma_f32_16x16x32_f16(false, a, false, bf,
                                                       (short)0, c,
                                                       false, false);
        }
        __syncthreads();
    }
#endif
    #pragma unroll
    for (int v = 0; v < 8; ++v) {
        int grow = row0 + v + 8 * hig;
        int gcol = col0 + n;
        size_t o = (size_t)grow * D_ + gcol;
        out[o] = c[v] + bo[gcol] + x[o];   // fused bias + residual
    }
}

extern "C" void kernel_launch(void* const* d_in, const int* in_sizes, int n_in,
                              void* d_out, int out_size, void* d_ws, size_t ws_size,
                              hipStream_t stream) {
    const float* x      = (const float*)d_in[0];
    const float* noise  = (const float*)d_in[1];
    const float* W1     = (const float*)d_in[2];
    const float* b1     = (const float*)d_in[3];
    const float* W2     = (const float*)d_in[4];
    const float* b2     = (const float*)d_in[5];
    const float* freqs  = (const float*)d_in[6];
    const float* amps   = (const float*)d_in[7];
    const float* phases = (const float*)d_in[8];
    const float* Wo     = (const float*)d_in[9];
    const float* bo     = (const float*)d_in[10];
    float* out = (float*)d_out;

    char* ws = (char*)d_ws;
    size_t off = 0;
    auto alloc = [&](size_t bytes) -> void* {
        void* p = ws + off;
        off += (bytes + 255) & ~(size_t)255;
        return p;
    };
    float*    mask   = (float*)alloc((size_t)H_ * M_ * S_ * 4);       // 4 MB
    float*    draw   = (float*)alloc((size_t)H_ * B_ * S_ * 4);       // 512 KB
    float*    pooled = (float*)alloc((size_t)B_ * D_ * 4);            // 16 KB
    float*    wts    = (float*)alloc((size_t)H_ * B_ * M_ * 4);       // 4 KB
    float*    dsum   = (float*)alloc((size_t)H_ * B_ * 4);            // 128 B
    _Float16* wofT   = (_Float16*)alloc((size_t)D_ * D_ * 2);         // 2 MB
    _Float16* mh     = (_Float16*)alloc((size_t)B_ * S_ * D_ * 2);    // 32 MB
    (void)ws_size; (void)in_sizes; (void)n_in; (void)out_size;

    k_zero    <<<16, 256, 0, stream>>>(pooled, dsum);
    k_pool    <<<dim3(16, 16), 256, 0, stream>>>(x, pooled);
    k_gate    <<<H_ * B_, 128, 0, stream>>>(pooled, W1, b1, W2, b2, wts);
    k_mask    <<<(H_ * M_ * S_) / 256, 256, 0, stream>>>(freqs, amps, phases, mask);
    k_density <<<dim3(S_ / 256, B_, H_), 256, 0, stream>>>(mask, wts, noise, draw, dsum);
    k_modulate<<<(B_ * S_ * D_) / 256, 256, 0, stream>>>(x, draw, dsum, mh);
    k_wo16t   <<<dim3(32, 32), 256, 0, stream>>>(Wo, wofT);
    k_gemm    <<<dim3(D_ / 128, (B_ * S_) / 16), 256, 0, stream>>>(mh, wofT, x, bo, out);
}
